// SwinTransformerBlock_10608569221174
// MI455X (gfx1250) — matrix operand order repaired and checked by reading, not compiled, not run on hardware
//
#include <hip/hip_runtime.h>
#include <hip/hip_bf16.h>
#include <stdint.h>

// ---------------- CDNA5 WMMA plumbing (wave32, 16x16x32 f16 -> f32) ----------------
typedef __attribute__((ext_vector_type(16))) _Float16 v16h;
typedef __attribute__((ext_vector_type(8)))  _Float16 v8h;
typedef __attribute__((ext_vector_type(8)))  float    v8f;

union V16 { v16h v; v8h h[2]; };

#define WMMA_F16(a, b, c) \
  __builtin_amdgcn_wmma_f32_16x16x32_f16(false, (a), false, (b), (short)0, (c), false, false)

// ---------------- problem constants ----------------
#define BATCH 2
#define DIM   28
#define Cdim  192
#define NHd   6
#define HD    32
#define WSz   7
#define SFT   3
#define NSEQ  343          // 7^3 tokens per window
#define NPAD  352          // 22*16
#define NWloc 64           // (28/7)^3 windows per batch
#define NWIN  128          // BATCH * NWloc
#define NTOK  43904        // BATCH * 28^3  (also GEMM M; = 686*64)
#define CH    768          // mlp hidden
#define NN2   117649       // 343*343

// A operand: 16 rows x 32 k, f16, row-major source with leading dim ld (halves).
// lane<16: row=lane, k = 0..7 & 16..23 ; lane>=16: row=lane-16, k = 8..15 & 24..31
__device__ __forceinline__ v16h ld_a_tile(const _Float16* __restrict__ base, int ld, int lane) {
  const _Float16* p = base + (size_t)(lane & 15) * ld + ((lane >> 4) << 3);
  V16 r; r.h[0] = *(const v8h*)p; r.h[1] = *(const v8h*)(p + 16);
  return r.v;
}
// B operand: 32 k x 16 cols; source is row-major [col][k] (i.e. W[j][k]).
// lane<16: col=lane, k = 0..7 & 16..23 ; lane>=16: col=lane-16, k = 8..15 & 24..31
__device__ __forceinline__ v16h ld_b_tile(const _Float16* __restrict__ bt, int ld, int lane) {
  const _Float16* p = bt + (size_t)(lane & 15) * ld + ((lane >> 4) << 3);
  V16 r; r.h[0] = *(const v8h*)p; r.h[1] = *(const v8h*)(p + 16);
  return r.v;
}

// 16 rows x 64 cols GEMM core per wave: acc[4] over K (multiple of 32).
// Bw must already be offset to column group n0 (i.e. Bw = W + n0*ldb).
__device__ __forceinline__ void gemm_rows16x64(const _Float16* __restrict__ Arow, int lda,
                                               const _Float16* __restrict__ Bw, int ldb, int K,
                                               v8f (&acc)[4], int lane) {
  for (int k0 = 0; k0 < K; k0 += 32) {
    v16h a = ld_a_tile(Arow + k0, lda, lane);
#pragma unroll
    for (int t = 0; t < 4; ++t) {
      v16h b = ld_b_tile(Bw + (size_t)(t * 16) * ldb + k0, ldb, lane);
      acc[t] = WMMA_F16(a, b, acc[t]);
    }
  }
}

// ---------------- utility kernels ----------------
__global__ void k_zero32(uint32_t* p, size_t n) {
  size_t i = (size_t)blockIdx.x * blockDim.x + threadIdx.x;
  size_t st = (size_t)gridDim.x * blockDim.x;
  for (; i < n; i += st) p[i] = 0u;
}

__global__ void k_f32tof16(const float* __restrict__ s, _Float16* __restrict__ d, int n) {
  int i = blockIdx.x * blockDim.x + threadIdx.x;
  if (i < n) d[i] = (_Float16)s[i];
}

// precompute relative-position bias per head: biasPre[h][n][m], n,m in [0,343)
__global__ void k_biaspre(const float* __restrict__ rpb, float* __restrict__ biasPre) {
  int id = blockIdx.x * blockDim.x + threadIdx.x;
  if (id >= NN2) return;
  int n = id / NSEQ, m = id % NSEQ;
  int a1 = n / 49, b1 = (n / 7) % 7, c1 = n % 7;
  int a2 = m / 49, b2 = (m / 7) % 7, c2 = m % 7;
  int idx = (a1 - a2 + 6) * 169 + (b1 - b2 + 6) * 13 + (c1 - c2 + 6);
#pragma unroll
  for (int h = 0; h < NHd; ++h) biasPre[(size_t)h * NN2 + id] = rpb[idx * NHd + h];
}

// LN1 + cyclic shift (-3,-3,-3) + window partition -> xw f16 [win*343+n][192]
__global__ __launch_bounds__(256) void k_ln1_shift(const float* __restrict__ x,
                                                   const float* __restrict__ g,
                                                   const float* __restrict__ bta,
                                                   _Float16* __restrict__ xw) {
  int wid = blockIdx.x * 8 + (threadIdx.x >> 5);  // token id
  int lane = threadIdx.x & 31;
  if (wid >= NTOK) return;
  const float* xp = x + (size_t)wid * Cdim;
  float v[6]; float s = 0.f;
#pragma unroll
  for (int i = 0; i < 6; ++i) { v[i] = xp[lane + i * 32]; s += v[i]; }
#pragma unroll
  for (int o = 16; o; o >>= 1) s += __shfl_xor(s, o, 32);
  float mean = s * (1.0f / Cdim);
  float vs = 0.f;
#pragma unroll
  for (int i = 0; i < 6; ++i) { float d = v[i] - mean; vs += d * d; }
#pragma unroll
  for (int o = 16; o; o >>= 1) vs += __shfl_xor(vs, o, 32);
  float inv = rsqrtf(vs * (1.0f / Cdim) + 1e-5f);

  int b = wid / (DIM * DIM * DIM);
  int l = wid % (DIM * DIM * DIM);
  int h = l / (DIM * DIM), w = (l / DIM) % DIM, t = l % DIM;
  int hs = (h + DIM - SFT) % DIM, ws2 = (w + DIM - SFT) % DIM, ts = (t + DIM - SFT) % DIM;
  int win = ((b * 4 + hs / WSz) * 4 + ws2 / WSz) * 4 + ts / WSz;
  int n = ((hs % WSz) * WSz + (ws2 % WSz)) * WSz + (ts % WSz);
  _Float16* op = xw + ((size_t)win * NSEQ + n) * Cdim;
#pragma unroll
  for (int i = 0; i < 6; ++i) {
    int c = lane + i * 32;
    op[c] = (_Float16)((v[i] - mean) * inv * g[c] + bta[c]);
  }
}

// plain LN2 over x1 (= d_out) -> h2 f16, same row order
__global__ __launch_bounds__(256) void k_ln2(const float* __restrict__ x,
                                             const float* __restrict__ g,
                                             const float* __restrict__ bta,
                                             _Float16* __restrict__ h2) {
  int wid = blockIdx.x * 8 + (threadIdx.x >> 5);
  int lane = threadIdx.x & 31;
  if (wid >= NTOK) return;
  const float* xp = x + (size_t)wid * Cdim;
  float v[6]; float s = 0.f;
#pragma unroll
  for (int i = 0; i < 6; ++i) { v[i] = xp[lane + i * 32]; s += v[i]; }
#pragma unroll
  for (int o = 16; o; o >>= 1) s += __shfl_xor(s, o, 32);
  float mean = s * (1.0f / Cdim);
  float vs = 0.f;
#pragma unroll
  for (int i = 0; i < 6; ++i) { float d = v[i] - mean; vs += d * d; }
#pragma unroll
  for (int o = 16; o; o >>= 1) vs += __shfl_xor(vs, o, 32);
  float inv = rsqrtf(vs * (1.0f / Cdim) + 1e-5f);
  _Float16* op = h2 + (size_t)wid * Cdim;
#pragma unroll
  for (int i = 0; i < 6; ++i) {
    int c = lane + i * 32;
    op[c] = (_Float16)((v[i] - mean) * inv * g[c] + bta[c]);
  }
}

// ---------------- GEMM kernels (WMMA) ----------------
// QKV: [43904x192] @ [576x192]^T, scatter epilogue to q(scaled)/k(padded)/vT
__global__ __launch_bounds__(128) void k_gemm_qkv(const _Float16* __restrict__ A,
                                                  const _Float16* __restrict__ W,
                                                  const float* __restrict__ bias,
                                                  _Float16* __restrict__ q,
                                                  _Float16* __restrict__ k,
                                                  _Float16* __restrict__ vt) {
  const int lane = threadIdx.x & 31;
  const int m0 = blockIdx.x * 64 + (threadIdx.x >> 5) * 16;
  const int n0 = blockIdx.y * 64;
  v8f acc[4] = {};
  gemm_rows16x64(A + (size_t)m0 * Cdim, Cdim, W + (size_t)n0 * Cdim, Cdim, Cdim, acc, lane);
  const int cl = lane & 15, rAdd = (lane >> 4) << 3;
#pragma unroll
  for (int t = 0; t < 4; ++t) {
    int j = n0 + t * 16 + cl;
    int sg = j / Cdim, hh = (j % Cdim) >> 5, d = j & 31;
#pragma unroll
    for (int r = 0; r < 8; ++r) {
      int m = m0 + r + rAdd;
      float val = acc[t][r] + bias[j];
      int win = m / NSEQ, n = m % NSEQ;
      size_t gh = (size_t)win * NHd + hh;
      if (sg == 0)      q[(gh * NSEQ + n) * HD + d]  = (_Float16)(val * 0.17677669529663687f);
      else if (sg == 1) k[(gh * NPAD + n) * HD + d]  = (_Float16)val;
      else              vt[(gh * HD + d) * NPAD + n] = (_Float16)val;
    }
  }
}

// proj: [43904x192] @ [192x192]^T; epilogue = window-reverse + roll(+3) + residual -> x1 (d_out)
__global__ __launch_bounds__(128) void k_gemm_proj(const _Float16* __restrict__ A,
                                                   const _Float16* __restrict__ W,
                                                   const float* __restrict__ bias,
                                                   const float* __restrict__ xin,
                                                   float* __restrict__ x1) {
  const int lane = threadIdx.x & 31;
  const int m0 = blockIdx.x * 64 + (threadIdx.x >> 5) * 16;
  const int n0 = blockIdx.y * 64;
  v8f acc[4] = {};
  gemm_rows16x64(A + (size_t)m0 * Cdim, Cdim, W + (size_t)n0 * Cdim, Cdim, Cdim, acc, lane);
  const int cl = lane & 15, rAdd = (lane >> 4) << 3;
#pragma unroll
  for (int t = 0; t < 4; ++t) {
    int j = n0 + t * 16 + cl;
#pragma unroll
    for (int r = 0; r < 8; ++r) {
      int m = m0 + r + rAdd;
      float val = acc[t][r] + bias[j];
      int win = m / NSEQ, n = m % NSEQ;
      int b = win >> 6, iw = win & 63;
      int a = n / 49, bq = (n / 7) % 7, cq = n % 7;
      int h = (iw >> 4) * WSz + a + SFT;        if (h >= DIM) h -= DIM;
      int w = (((iw >> 2) & 3)) * WSz + bq + SFT; if (w >= DIM) w -= DIM;
      int tt = (iw & 3) * WSz + cq + SFT;       if (tt >= DIM) tt -= DIM;
      size_t idx = (((size_t)b * (DIM * DIM * DIM)) + ((h * DIM + w) * DIM + tt)) * Cdim + j;
      x1[idx] = xin[idx] + val;
    }
  }
}

// fc1: [43904x192] @ [768x192]^T, exact GELU epilogue -> mbuf f16
__global__ __launch_bounds__(128) void k_gemm_fc1(const _Float16* __restrict__ A,
                                                  const _Float16* __restrict__ W,
                                                  const float* __restrict__ bias,
                                                  _Float16* __restrict__ mb) {
  const int lane = threadIdx.x & 31;
  const int m0 = blockIdx.x * 64 + (threadIdx.x >> 5) * 16;
  const int n0 = blockIdx.y * 64;
  v8f acc[4] = {};
  gemm_rows16x64(A + (size_t)m0 * Cdim, Cdim, W + (size_t)n0 * Cdim, Cdim, Cdim, acc, lane);
  const int cl = lane & 15, rAdd = (lane >> 4) << 3;
#pragma unroll
  for (int t = 0; t < 4; ++t) {
    int j = n0 + t * 16 + cl;
#pragma unroll
    for (int r = 0; r < 8; ++r) {
      int m = m0 + r + rAdd;
      float v = acc[t][r] + bias[j];
      float ge = 0.5f * v * (1.0f + erff(v * 0.70710678118654752f));
      mb[(size_t)m * CH + j] = (_Float16)ge;
    }
  }
}

// fc2: [43904x768] @ [192x768]^T, epilogue out = x1 + val (in-place on d_out)
__global__ __launch_bounds__(128) void k_gemm_fc2(const _Float16* __restrict__ A,
                                                  const _Float16* __restrict__ W,
                                                  const float* __restrict__ bias,
                                                  float* __restrict__ out) {
  const int lane = threadIdx.x & 31;
  const int m0 = blockIdx.x * 64 + (threadIdx.x >> 5) * 16;
  const int n0 = blockIdx.y * 64;
  v8f acc[4] = {};
  gemm_rows16x64(A + (size_t)m0 * CH, CH, W + (size_t)n0 * CH, CH, CH, acc, lane);
  const int cl = lane & 15, rAdd = (lane >> 4) << 3;
#pragma unroll
  for (int t = 0; t < 4; ++t) {
    int j = n0 + t * 16 + cl;
#pragma unroll
    for (int r = 0; r < 8; ++r) {
      int m = m0 + r + rAdd;
      size_t idx = (size_t)m * Cdim + j;
      out[idx] = out[idx] + acc[t][r] + bias[j];
    }
  }
}

// ---------------- fused windowed attention ----------------
// grid: (NWIN*NHd)*6 blocks; block = 128 threads (4 waves); each wave owns 16 query rows,
// full 352-col score strip in 22 WMMA accumulators (hd=32 -> 1 WMMA per score tile).
__global__ __launch_bounds__(128, 1) void k_attn(const _Float16* __restrict__ qbuf,
                                                 const _Float16* __restrict__ kbuf,
                                                 const _Float16* __restrict__ vtbuf,
                                                 const float* __restrict__ biasPre,
                                                 const float* __restrict__ mask,
                                                 _Float16* __restrict__ obuf) {
  __shared__ __align__(16) _Float16 Plds[4][16 * NPAD];
  const int lane = threadIdx.x & 31;
  const int wv = threadIdx.x >> 5;
  const int g = blockIdx.x / 6;          // win*NHd + head
  const int chunk = blockIdx.x % 6;
  const int win = g / NHd, hh = g % NHd;
  const int iw = win & 63;
  const int base_row = chunk * 64 + wv * 16;
  const int cl = lane & 15;
  const int rAdd = (lane >> 4) << 3;

  // Q tile (A operand), rows clamped into range
  int ra = base_row + cl; if (ra > NSEQ - 1) ra = NSEQ - 1;
  const _Float16* qp = qbuf + ((size_t)g * NSEQ + ra) * HD + ((lane >> 4) << 3);
  V16 av; av.h[0] = *(const v8h*)qp; av.h[1] = *(const v8h*)(qp + 16);

  // S = q @ k^T over 22 column tiles
  v8f S[22];
#pragma unroll
  for (int t = 0; t < 22; ++t) {
    const _Float16* kp = kbuf + ((size_t)g * NPAD + t * 16 + cl) * HD + ((lane >> 4) << 3);
    V16 bv; bv.h[0] = *(const v8h*)kp; bv.h[1] = *(const v8h*)(kp + 16);
    v8f z = {};
    S[t] = WMMA_F16(av.v, bv.v, z);
  }

  // bias + mask, row max
  const float* bp = biasPre + (size_t)hh * NN2;
  const float* mp = mask + (size_t)iw * NN2;
  float rmax[8], rsum[8];
#pragma unroll
  for (int r = 0; r < 8; ++r) rmax[r] = -3.0e38f;
#pragma unroll
  for (int t = 0; t < 22; ++t) {
    int col = t * 16 + cl;
#pragma unroll
    for (int r = 0; r < 8; ++r) {
      int rowg = base_row + r + rAdd;
      float v;
      if (col < NSEQ && rowg < NSEQ) {
        int e = rowg * NSEQ + col;
        v = S[t][r] + bp[e] + mp[e];
      } else v = -3.0e38f;
      S[t][r] = v;
      rmax[r] = fmaxf(rmax[r], v);
    }
  }
#pragma unroll
  for (int r = 0; r < 8; ++r) {
    float m = rmax[r];
    m = fmaxf(m, __shfl_xor(m, 1, 32));
    m = fmaxf(m, __shfl_xor(m, 2, 32));
    m = fmaxf(m, __shfl_xor(m, 4, 32));
    m = fmaxf(m, __shfl_xor(m, 8, 32));
    rmax[r] = m; rsum[r] = 0.f;
  }

  // exp, accumulate row sums, stash unnormalized P (f16) into this wave's LDS slice
  _Float16* pw = &Plds[wv][0];
#pragma unroll
  for (int t = 0; t < 22; ++t) {
    int col = t * 16 + cl;
#pragma unroll
    for (int r = 0; r < 8; ++r) {
      float p = __expf(S[t][r] - rmax[r]);
      rsum[r] += p;
      pw[(r + rAdd) * NPAD + col] = (_Float16)p;
    }
  }
#pragma unroll
  for (int r = 0; r < 8; ++r) {
    float s = rsum[r];
    s += __shfl_xor(s, 1, 32);
    s += __shfl_xor(s, 2, 32);
    s += __shfl_xor(s, 4, 32);
    s += __shfl_xor(s, 8, 32);
    rsum[r] = s;
  }

  // O = P @ V via 11 K-steps; B operand from transposed V (row load)
  v8f O0 = {}, O1 = {};
#pragma unroll
  for (int kc = 0; kc < 11; ++kc) {
    const _Float16* ap = pw + (size_t)cl * NPAD + kc * 32 + ((lane >> 4) << 3);
    V16 a2; a2.h[0] = *(const v8h*)ap; a2.h[1] = *(const v8h*)(ap + 16);
    const _Float16* b0 = vtbuf + ((size_t)g * HD + cl) * NPAD + kc * 32 + ((lane >> 4) << 3);
    V16 bb0; bb0.h[0] = *(const v8h*)b0; bb0.h[1] = *(const v8h*)(b0 + 16);
    O0 = WMMA_F16(a2.v, bb0.v, O0);
    const _Float16* b1 = vtbuf + ((size_t)g * HD + 16 + cl) * NPAD + kc * 32 + ((lane >> 4) << 3);
    V16 bb1; bb1.h[0] = *(const v8h*)b1; bb1.h[1] = *(const v8h*)(b1 + 16);
    O1 = WMMA_F16(a2.v, bb1.v, O1);
  }

  // normalize and write (Bn, N, C) with C index h*32+d
  size_t obase = (size_t)win * NSEQ * Cdim + (size_t)hh * HD;
#pragma unroll
  for (int r = 0; r < 8; ++r) {
    int rowg = base_row + r + rAdd;
    if (rowg < NSEQ) {
      float inv = 1.0f / rsum[r];
      obuf[obase + (size_t)rowg * Cdim + cl]      = (_Float16)(O0[r] * inv);
      obuf[obase + (size_t)rowg * Cdim + 16 + cl] = (_Float16)(O1[r] * inv);
    }
  }
}

// ---------------- launcher ----------------
extern "C" void kernel_launch(void* const* d_in, const int* in_sizes, int n_in,
                              void* d_out, int out_size, void* d_ws, size_t ws_size,
                              hipStream_t stream) {
  (void)in_sizes; (void)n_in; (void)out_size; (void)ws_size;
  const float* x      = (const float*)d_in[0];
  const float* mask   = (const float*)d_in[1];
  const float* gamma1 = (const float*)d_in[2];
  const float* beta1  = (const float*)d_in[3];
  const float* qkv_w  = (const float*)d_in[4];
  const float* qkv_b  = (const float*)d_in[5];
  const float* rpb    = (const float*)d_in[6];
  const float* proj_w = (const float*)d_in[7];
  const float* proj_b = (const float*)d_in[8];
  const float* gamma2 = (const float*)d_in[9];
  const float* beta2  = (const float*)d_in[10];
  const float* fc1_w  = (const float*)d_in[11];
  const float* fc1_b  = (const float*)d_in[12];
  const float* fc2_w  = (const float*)d_in[13];
  const float* fc2_b  = (const float*)d_in[14];
  float* out = (float*)d_out;            // doubles as x1 residual buffer

  // ---- workspace layout (total ~105 MB; aliased lifetimes) ----
  auto al = [](size_t v) { return (v + 255) & ~(size_t)255; };
  char* ws = (char*)d_ws;
  size_t o_w    = 0;                                  // weights f16 (persistent)
  size_t sz_w   = (size_t)(576*192 + 192*192 + 768*192 + 192*768) * 2;
  size_t o_bias = al(o_w + sz_w);                     // biasPre f32 [6][343][343]
  size_t o_xw   = al(o_bias + (size_t)NHd * NN2 * 4); // xw f16 ; later aliased by obuf
  size_t o_q    = al(o_xw + (size_t)NTOK * Cdim * 2); // q f16  ; later aliased by h2
  size_t o_k    = al(o_q + (size_t)NWIN * NHd * NSEQ * HD * 2); // k f16 ; later aliased by mbuf
  size_t o_vt   = al(o_k + (size_t)NWIN * NHd * NPAD * HD * 2); // vT f16
  size_t sz_kvt = (o_vt + (size_t)NWIN * NHd * NPAD * HD * 2) - o_k;

  _Float16* wqkv = (_Float16*)(ws + o_w);
  _Float16* wprj = wqkv + 576 * 192;
  _Float16* wfc1 = wprj + 192 * 192;
  _Float16* wfc2 = wfc1 + 768 * 192;
  float*    biasPre = (float*)(ws + o_bias);
  _Float16* xw   = (_Float16*)(ws + o_xw);
  _Float16* obuf = (_Float16*)(ws + o_xw);   // alias: xw dead after QKV GEMM
  _Float16* qb   = (_Float16*)(ws + o_q);
  _Float16* h2   = (_Float16*)(ws + o_q);    // alias: q dead after attention
  _Float16* kb   = (_Float16*)(ws + o_k);
  _Float16* vtb  = (_Float16*)(ws + o_vt);
  _Float16* mbuf = (_Float16*)(ws + o_k);    // alias: k/vT dead after attention (67.4MB)

  // 0. zero k/vT (covers the 343->352 token padding)
  k_zero32<<<2048, 256, 0, stream>>>((uint32_t*)(ws + o_k), sz_kvt / 4);
  // 1. weights -> f16
  k_f32tof16<<<(576*192 + 255) / 256, 256, 0, stream>>>(qkv_w, wqkv, 576 * 192);
  k_f32tof16<<<(192*192 + 255) / 256, 256, 0, stream>>>(proj_w, wprj, 192 * 192);
  k_f32tof16<<<(768*192 + 255) / 256, 256, 0, stream>>>(fc1_w, wfc1, 768 * 192);
  k_f32tof16<<<(192*768 + 255) / 256, 256, 0, stream>>>(fc2_w, wfc2, 192 * 768);
  // 2. relative position bias table expansion
  k_biaspre<<<(NN2 + 255) / 256, 256, 0, stream>>>(rpb, biasPre);
  // 3. LN1 + shift + window partition
  k_ln1_shift<<<NTOK / 8, 256, 0, stream>>>(x, gamma1, beta1, xw);
  // 4. QKV GEMM (M=43904, N=576, K=192) with q/k/vT scatter
  k_gemm_qkv<<<dim3(NTOK / 64, 576 / 64), 128, 0, stream>>>(xw, wqkv, qkv_b, qb, kb, vtb);
  // 5. windowed attention (768 window-heads x 6 row chunks)
  k_attn<<<NWIN * NHd * 6, 128, 0, stream>>>(qb, kb, vtb, biasPre, mask, obuf);
  // 6. proj GEMM + reverse/unshift + residual -> x1 (stored in d_out)
  k_gemm_proj<<<dim3(NTOK / 64, 192 / 64), 128, 0, stream>>>(obuf, wprj, proj_b, x, out);
  // 7. LN2
  k_ln2<<<NTOK / 8, 256, 0, stream>>>(out, gamma2, beta2, h2);
  // 8. fc1 + GELU
  k_gemm_fc1<<<dim3(NTOK / 64, 768 / 64), 128, 0, stream>>>(h2, wfc1, fc1_b, mbuf);
  // 9. fc2 + residual (in-place on d_out)
  k_gemm_fc2<<<dim3(NTOK / 64, 192 / 64), 128, 0, stream>>>(mbuf, wfc2, fc2_b, out);
}
